// ProbSparseSelfAttention_90374701842687
// MI455X (gfx1250) — matrix operand order repaired and checked by reading, not compiled, not run on hardware
//
#include <hip/hip_runtime.h>
#include <hip/hip_bf16.h>
#include <math.h>
#include <stdint.h>

// ---------------- problem constants ----------------
constexpr int kB  = 2;
constexpr int kL  = 2048;
constexpr int kDM = 1024;
constexpr int kH  = 16;
constexpr int kHD = 64;               // head dim
constexpr int kU  = 409;              // max(1, L // TOP_FACTOR)
constexpr int kQT = (kU + 15) / 16;   // 26 query tiles of 16
constexpr int kM  = kB * kL;          // 4096 rows for the big GEMMs

typedef __attribute__((ext_vector_type(16))) __bf16        v16bf;
typedef __attribute__((ext_vector_type(8)))  float         v8f;
typedef __attribute__((ext_vector_type(8)))  unsigned int  v8u;
typedef __attribute__((ext_vector_type(4)))  unsigned int  v4u;
typedef __attribute__((ext_vector_type(8)))  int           v8i;
typedef __attribute__((ext_vector_type(4)))  int           v4i;

// ---- fp32 -> bf16 pair packing: 2x v_add + 1x v_perm_b32 ----
// round-to-nearest(ties-away); perm selector picks a[3:2] then b[3:2]
__device__ __forceinline__ unsigned pk(float x, float y) {
    unsigned a = __builtin_bit_cast(unsigned, x) + 0x8000u;
    unsigned b = __builtin_bit_cast(unsigned, y) + 0x8000u;
    return __builtin_amdgcn_perm(b, a, 0x07060302u);
}
__device__ __forceinline__ unsigned short f2bf_bits(float f) {
    return (unsigned short)((__builtin_bit_cast(unsigned, f) + 0x8000u) >> 16);
}
// v16bf from 4 float4 chunks (element e lives in p[e/2] half e%2)
__device__ __forceinline__ v16bf mk16(float4 a, float4 b, float4 c, float4 d) {
    v8u p;
    p[0] = pk(a.x, a.y); p[1] = pk(a.z, a.w);
    p[2] = pk(b.x, b.y); p[3] = pk(b.z, b.w);
    p[4] = pk(c.x, c.y); p[5] = pk(c.z, c.w);
    p[6] = pk(d.x, d.y); p[7] = pk(d.z, d.w);
    return __builtin_bit_cast(v16bf, p);
}

// =====================================================================
// Tensor Data Mover: load a 64x32 fp32 tile (row stride = stride_elts)
// into LDS at byte offset lds_off. Descriptor per CDNA5 ISA ch.8:
//  g0: [1:0]=count=1 | [63:32]=lds_addr | [120:64]=global_addr | type=2
//  g1: [17:16]=data_size(4B) | tensor_dim0/1 | tile_dim0/1 | dim0_stride
// 2D tile -> groups 2/3 zero (tile_dim2..4 = 0 -> unused).
// =====================================================================
#if __has_builtin(__builtin_amdgcn_tensor_load_to_lds)
#define HAVE_TDM 1
__device__ __forceinline__ void tdm_load_tile_64x32(
    unsigned lds_off, const float* gsrc, unsigned stride_elts)
{
    unsigned long long ga = (unsigned long long)(uintptr_t)gsrc;
    v4u g0;
    g0[0] = 1u;                                        // count=1 (user D#)
    g0[1] = lds_off;                                   // lds_addr bytes
    g0[2] = (unsigned)(ga & 0xFFFFFFFFu);              // global_addr[31:0]
    g0[3] = (unsigned)((ga >> 32) & 0x01FFFFFFu) | (2u << 30); // [56:32]|type=2
    v8i g1;
    g1[0] = (int)(2u << 16);          // wg_mask=0, data_size=2 (4 bytes)
    g1[1] = (int)(32u << 16);         // tensor_dim0 = 32 (lo16 at [31:16])
    g1[2] = (int)(64u << 16);         // dim0_hi=0 | tensor_dim1 = 64
    g1[3] = (int)(32u << 16);         // dim1_hi=0 | tile_dim0 = 32
    g1[4] = (int)64u;                 // tile_dim1 = 64 | tile_dim2 = 0
    g1[5] = (int)stride_elts;         // tensor_dim0_stride[31:0]
    g1[6] = 0;                        // stride0_hi | stride1_lo (2D: unused)
    g1[7] = 0;
    v4i z4 = {0, 0, 0, 0};
#if defined(__clang_major__) && (__clang_major__ >= 23)
    v8i z8 = {0, 0, 0, 0, 0, 0, 0, 0};
    __builtin_amdgcn_tensor_load_to_lds(g0, g1, z4, z4, z8, 0);
#else
    __builtin_amdgcn_tensor_load_to_lds(g0, g1, z4, z4, 0);
#endif
}
#else
#define HAVE_TDM 0
#endif

// =====================================================================
// GEMM:  C[M,N] = A[M,K] * W[N,K]^T + bias[N]
// Block: 256 thr = 8 waves, 64x64 C-tile. A(64x32) & W(64x32) k-slices
// double-buffered in LDS via TDM: wave0 issues stage i+1's two tile
// loads, then s_wait_tensorcnt 2 (in-order completion => stage i done),
// so the DMA of the next slice overlaps this slice's WMMAs.
// Each wave: one A-frag, two W-frags -> two wmma per k-step.
// Frag layouts (A: lane=row, 16 contiguous K at 16*hi; B: lane=col,
// 16 contiguous K at 16*hi; C: VGPR v -> row v+8*hi, col=lane&15).
// =====================================================================
__global__ __launch_bounds__(256) void gemm_bias_wmma(
    const float* __restrict__ A, const float* __restrict__ W,
    const float* __restrict__ bias, float* __restrict__ C,
    int M, int N, int K)
{
    __shared__ float As[2][64][32];
    __shared__ float Wsm[2][64][32];

    const int lane = threadIdx.x & 31;
    const int wave = threadIdx.x >> 5;
    const int r  = lane & 15;
    const int hi = lane >> 4;
    const int mloc = (wave & 3) * 16;        // 4 row tiles
    const int nloc = (wave >> 2) * 32;       // 2 col pairs
    const int mblk = blockIdx.x * 64;
    const int nblk = blockIdx.y * 64;

    v8f acc0 = {}, acc1 = {};

#if HAVE_TDM
    const unsigned ldsA0 = (unsigned)(uintptr_t)&As[0][0][0];
    const unsigned ldsA1 = (unsigned)(uintptr_t)&As[1][0][0];
    const unsigned ldsW0 = (unsigned)(uintptr_t)&Wsm[0][0][0];
    const unsigned ldsW1 = (unsigned)(uintptr_t)&Wsm[1][0][0];
    const bool issuer = (__builtin_amdgcn_readfirstlane((int)threadIdx.x) == 0);
    if (issuer) {                           // prime stage 0 into buffer 0
        tdm_load_tile_64x32(ldsA0, A + (size_t)mblk * K, (unsigned)K);
        tdm_load_tile_64x32(ldsW0, W + (size_t)nblk * K, (unsigned)K);
    }
#endif

    for (int k0 = 0; k0 < K; k0 += 32) {
        const int cur = (k0 >> 5) & 1;
#if HAVE_TDM
        if (issuer) {
            if (k0 + 32 < K) {              // prefetch stage i+1 into other buf
                tdm_load_tile_64x32(cur ? ldsA0 : ldsA1,
                                    A + (size_t)mblk * K + k0 + 32, (unsigned)K);
                tdm_load_tile_64x32(cur ? ldsW0 : ldsW1,
                                    W + (size_t)nblk * K + k0 + 32, (unsigned)K);
                __builtin_amdgcn_s_wait_tensorcnt(2);  // stage i complete
            } else {
                __builtin_amdgcn_s_wait_tensorcnt(0);
            }
        }
#else
        for (int idx = threadIdx.x; idx < 64 * 32 / 4; idx += 256) {
            const int row = idx >> 3, c4 = (idx & 7) * 4;
            ((float4*)As[cur])[idx]  = *(const float4*)(A + (size_t)(mblk + row) * K + k0 + c4);
            ((float4*)Wsm[cur])[idx] = *(const float4*)(W + (size_t)(nblk + row) * K + k0 + c4);
        }
#endif
        __syncthreads();                    // stage i visible to all waves

        const float4* ar = (const float4*)&As[cur][mloc + r][0];
        const v16bf af = mk16(ar[4 * hi], ar[4 * hi + 1],           // K 16*hi..+8
                              ar[4 * hi + 2], ar[4 * hi + 3]);      // K 16*hi+8..+8

        const float4* w0 = (const float4*)&Wsm[cur][nloc + r][0];
        const float4* w1 = (const float4*)&Wsm[cur][nloc + 16 + r][0];
        const v16bf bf0 = mk16(w0[4 * hi], w0[4 * hi + 1], w0[4 * hi + 2], w0[4 * hi + 3]);
        const v16bf bf1 = mk16(w1[4 * hi], w1[4 * hi + 1], w1[4 * hi + 2], w1[4 * hi + 3]);

        acc0 = __builtin_amdgcn_wmma_f32_16x16x32_bf16(
            false, af, false, bf0, (short)0, acc0, false, false);
        acc1 = __builtin_amdgcn_wmma_f32_16x16x32_bf16(
            false, af, false, bf1, (short)0, acc1, false, false);

        __syncthreads();                    // reads done before buffer reuse
    }

    const int c0 = nblk + nloc + r;
    const float b0 = bias[c0];
    const float b1 = bias[c0 + 16];
#pragma unroll
    for (int v = 0; v < 8; ++v) {
        const int row = mblk + mloc + v + 8 * hi;
        C[(size_t)row * N + c0]      = acc0[v] + b0;
        C[(size_t)row * N + c0 + 16] = acc1[v] + b1;
    }
}

// =====================================================================
// kmean[b,h,d] = (1/L) * sum_l K[b,h,l,d] / ||K[b,h,l,:]||
// (exact linear collapse of mean_k of normalized-QK scores)
// =====================================================================
__global__ __launch_bounds__(256) void kmean_kernel(
    const float* __restrict__ Kb, float* __restrict__ kmean)
{
    __shared__ float invn[kL];
    __shared__ float kacc[kHD];
    const int bh = blockIdx.x;
    const int b = bh / kH, h = bh % kH;
    const int tid = threadIdx.x;
    const float* __restrict__ base = Kb + (size_t)b * kL * kDM + h * kHD;

    for (int l = tid; l < kL; l += 256) {
        const float* kp = base + (size_t)l * kDM;
        float s = 0.f;
#pragma unroll
        for (int d = 0; d < kHD; d += 4) {
            float4 v = *(const float4*)(kp + d);
            s += v.x * v.x + v.y * v.y + v.z * v.z + v.w * v.w;
        }
        invn[l] = rsqrtf(s);
    }
    if (tid < kHD) kacc[tid] = 0.f;
    __syncthreads();

    const int d = tid & 63;
    const int g = tid >> 6;
    float p = 0.f;
    for (int l = g * 512; l < (g + 1) * 512; ++l)
        p += base[(size_t)l * kDM + d] * invn[l];
    atomicAdd(&kacc[d], p);
    __syncthreads();

    if (tid < kHD) kmean[bh * kHD + tid] = kacc[tid] * (1.0f / kL);
}

// =====================================================================
// mean_scores[b,h,q] = (q / ||q||) . kmean[b,h,:]
// =====================================================================
__global__ __launch_bounds__(256) void mscore_kernel(
    const float* __restrict__ Qb, const float* __restrict__ kmean,
    float* __restrict__ msc)
{
    const int idx = blockIdx.x * 256 + threadIdx.x;
    if (idx >= kB * kH * kL) return;
    const int q = idx % kL;
    const int bh = idx / kL;
    const int b = bh / kH, h = bh % kH;
    const float* __restrict__ qp = Qb + ((size_t)(b * kL + q)) * kDM + h * kHD;
    const float* __restrict__ km = kmean + bh * kHD;
    float ss = 0.f, dot = 0.f;
#pragma unroll
    for (int d = 0; d < kHD; d += 4) {
        float4 v = *(const float4*)(qp + d);
        float4 m = *(const float4*)(km + d);
        ss  += v.x * v.x + v.y * v.y + v.z * v.z + v.w * v.w;
        dot += v.x * m.x + v.y * m.y + v.z * m.z + v.w * m.w;
    }
    msc[idx] = dot * rsqrtf(ss);
}

// =====================================================================
// Exact stable top-k by rank counting (rank<U -> slot=rank; no dups)
// =====================================================================
__global__ __launch_bounds__(256) void topk_kernel(
    const float* __restrict__ msc, int* __restrict__ top_idx)
{
    __shared__ float s[kL];
    const int bh = blockIdx.x;
    const int tid = threadIdx.x;
    for (int l = tid; l < kL; l += 256) s[l] = msc[bh * kL + l];
    __syncthreads();
    for (int q = tid; q < kL; q += 256) {
        const float my = s[q];
        int cnt = 0;
        for (int j = 0; j < kL; ++j) {
            const float o = s[j];
            cnt += (o > my) || (o == my && j < q);
        }
        if (cnt < kU) top_idx[bh * kU + cnt] = q;
    }
}

__global__ void zero_kernel(float4* __restrict__ p, long n4) {
    const long i = (long)blockIdx.x * blockDim.x + threadIdx.x;
    if (i < n4) p[i] = make_float4(0.f, 0.f, 0.f, 0.f);
}

// =====================================================================
// Sparse attention, flash style; one wave per (b, h, 16-query tile).
// S = Qsel*K^T (2 wmma k-steps x 2 key halves), online softmax with
// 16-lane shfl reductions, P through LDS (C-layout -> A-layout),
// ctx += P*V (4 wmma). Rows scatter into zeroed ctx via top_idx.
// =====================================================================
__global__ __launch_bounds__(32) void sparse_attn_kernel(
    const float* __restrict__ Qb, const float* __restrict__ Kb,
    const float* __restrict__ Vb, const int* __restrict__ top_idx,
    float* __restrict__ ctx)
{
    __shared__ unsigned short pbuf[16 * 32];

    const int lane = threadIdx.x;
    const int r  = lane & 15;
    const int hi = lane >> 4;
    const int bh = blockIdx.x / kQT;
    const int qt = blockIdx.x % kQT;
    const int b = bh / kH, h = bh % kH;
    const float scale = 0.125f;           // 1/sqrt(64)

    const int qrow = qt * 16 + r;
    const int qsel = (qrow < kU) ? top_idx[bh * kU + qrow] : 0;
    const float* __restrict__ qp = Qb + ((size_t)(b * kL + qsel)) * kDM + h * kHD;

    v16bf aq[2];
#pragma unroll
    for (int st = 0; st < 2; ++st) {
        const float4* c1 = (const float4*)(qp + st * 32 + 16 * hi);
        const float4* c2 = (const float4*)(qp + st * 32 + 16 * hi + 8);
        aq[st] = mk16(c1[0], c1[1], c2[0], c2[1]);
    }

    const float* __restrict__ kbase = Kb + (size_t)b * kL * kDM + h * kHD;
    const float* __restrict__ vbase = Vb + (size_t)b * kL * kDM + h * kHD;

    v8f o[4] = {v8f{}, v8f{}, v8f{}, v8f{}};
    float mrow[8], lrow[8];
#pragma unroll
    for (int v = 0; v < 8; ++v) { mrow[v] = -INFINITY; lrow[v] = 0.f; }

    for (int kb = 0; kb < kL; kb += 32) {
        v8f s0 = {}, s1 = {};
#pragma unroll
        for (int st = 0; st < 2; ++st) {
            const float4* p0 = (const float4*)(kbase + (size_t)(kb + r) * kDM + st * 32 + 16 * hi);
            const float4* p1 = (const float4*)(kbase + (size_t)(kb + 16 + r) * kDM + st * 32 + 16 * hi);
            const v16bf bk0 = mk16(p0[0], p0[1], p0[2], p0[3]);
            const v16bf bk1 = mk16(p1[0], p1[1], p1[2], p1[3]);
            s0 = __builtin_amdgcn_wmma_f32_16x16x32_bf16(
                false, aq[st], false, bk0, (short)0, s0, false, false);
            s1 = __builtin_amdgcn_wmma_f32_16x16x32_bf16(
                false, aq[st], false, bk1, (short)0, s1, false, false);
        }

#pragma unroll
        for (int v = 0; v < 8; ++v) {
            const float a0 = s0[v] * scale;
            const float a1 = s1[v] * scale;
            float t = fmaxf(a0, a1);
#pragma unroll
            for (int off = 1; off < 16; off <<= 1)
                t = fmaxf(t, __shfl_xor(t, off, 32));
            const float mn = fmaxf(mrow[v], t);
            const float sc = __expf(mrow[v] - mn);
            const float p0e = __expf(a0 - mn);
            const float p1e = __expf(a1 - mn);
            float rs = p0e + p1e;
#pragma unroll
            for (int off = 1; off < 16; off <<= 1)
                rs += __shfl_xor(rs, off, 32);
            lrow[v] = lrow[v] * sc + rs;
            mrow[v] = mn;
            o[0][v] *= sc; o[1][v] *= sc; o[2][v] *= sc; o[3][v] *= sc;

            const int row = v + 8 * hi;
            pbuf[row * 32 + r]      = f2bf_bits(p0e);
            pbuf[row * 32 + 16 + r] = f2bf_bits(p1e);
        }

        // P in A-layout: u32-pair reads (chunks at even ushort offsets)
        const unsigned* pb32 = (const unsigned*)pbuf;
        v8u pp;
#pragma unroll
        for (int j = 0; j < 4; ++j) {
            pp[j]     = pb32[(r * 32 + 16 * hi) / 2 + j];
            pp[4 + j] = pb32[(r * 32 + 16 * hi + 8) / 2 + j];
        }
        const v16bf pa = __builtin_bit_cast(v16bf, pp);

#pragma unroll
        for (int nt = 0; nt < 4; ++nt) {
            v8u bp;
#pragma unroll
            for (int j = 0; j < 8; ++j) {
                const float e0 = vbase[(size_t)(kb + 16 * hi + 2 * j) * kDM + nt * 16 + r];
                const float e1 = vbase[(size_t)(kb + 16 * hi + 2 * j + 1) * kDM + nt * 16 + r];
                bp[j] = pk(e0, e1);
            }
            const v16bf bv = __builtin_bit_cast(v16bf, bp);
            o[nt] = __builtin_amdgcn_wmma_f32_16x16x32_bf16(
                false, pa, false, bv, (short)0, o[nt], false, false);
        }
    }

#pragma unroll
    for (int v = 0; v < 8; ++v) {
        const int row = qt * 16 + v + 8 * hi;
        if (row < kU) {
            const int q = top_idx[bh * kU + row];
            const float inv = 1.0f / lrow[v];
            float* op = ctx + ((size_t)(b * kL + q)) * kDM + h * kHD + r;
            op[0]  = o[0][v] * inv;
            op[16] = o[1][v] * inv;
            op[32] = o[2][v] * inv;
            op[48] = o[3][v] * inv;
        }
    }
}

// =====================================================================
extern "C" void kernel_launch(void* const* d_in, const int* in_sizes, int n_in,
                              void* d_out, int out_size, void* d_ws, size_t ws_size,
                              hipStream_t stream) {
    (void)in_sizes; (void)n_in; (void)out_size; (void)ws_size;

    const float* queries = (const float*)d_in[0];
    const float* keys    = (const float*)d_in[1];
    const float* values  = (const float*)d_in[2];
    const float* Wq = (const float*)d_in[3];  const float* bq = (const float*)d_in[4];
    const float* Wk = (const float*)d_in[5];  const float* bk = (const float*)d_in[6];
    const float* Wv = (const float*)d_in[7];  const float* bv = (const float*)d_in[8];
    const float* Wo = (const float*)d_in[9];  const float* bo = (const float*)d_in[10];
    float* out = (float*)d_out;

    float* ws = (float*)d_ws;
    const size_t MN = (size_t)kM * kDM;
    float* Qb    = ws;
    float* Kb    = Qb + MN;
    float* Vb    = Kb + MN;
    float* ctx   = Vb + MN;
    float* kmean = ctx + MN;
    float* msc   = kmean + (size_t)kB * kH * kHD;
    int*   tidx  = (int*)(msc + (size_t)kB * kH * kL);

    dim3 gg(kM / 64, kDM / 64);

    gemm_bias_wmma<<<gg, 256, 0, stream>>>(queries, Wq, bq, Qb, kM, kDM, kDM);
    gemm_bias_wmma<<<gg, 256, 0, stream>>>(keys,    Wk, bk, Kb, kM, kDM, kDM);
    gemm_bias_wmma<<<gg, 256, 0, stream>>>(values,  Wv, bv, Vb, kM, kDM, kDM);

    kmean_kernel<<<kB * kH, 256, 0, stream>>>(Kb, kmean);
    mscore_kernel<<<(kB * kH * kL + 255) / 256, 256, 0, stream>>>(Qb, kmean, msc);
    topk_kernel<<<kB * kH, 256, 0, stream>>>(msc, tidx);

    zero_kernel<<<(int)((MN / 4 + 255) / 256), 256, 0, stream>>>((float4*)ctx, (long)(MN / 4));
    sparse_attn_kernel<<<kB * kH * kQT, 32, 0, stream>>>(Qb, Kb, Vb, tidx, ctx);

    gemm_bias_wmma<<<gg, 256, 0, stream>>>(ctx, Wo, bo, out, kM, kDM, kDM);
}
